// TernaryMLP_28080496181414
// MI455X (gfx1250) — compile-verified, matrix operand order checked by reading, not run on hardware
//
#include <hip/hip_runtime.h>
#include <stdint.h>

typedef __bf16 bf16_t;
typedef __attribute__((ext_vector_type(16))) __bf16 v16bf;
typedef __attribute__((ext_vector_type(8)))  float  v8f;

union Frag { v16bf v; uint4 q[2]; };

// ---------------- conversion kernels ----------------

__global__ __launch_bounds__(256)
void cvt_x_bf16(const float* __restrict__ in, bf16_t* __restrict__ out, long n) {
  long i = ((long)blockIdx.x * 256 + threadIdx.x) * 4;
  if (i >= n) return;
  float4 f = *(const float4*)(in + i);
  union { bf16_t b[4]; uint2 u; } o;
  o.b[0] = (bf16_t)f.x; o.b[1] = (bf16_t)f.y;
  o.b[2] = (bf16_t)f.z; o.b[3] = (bf16_t)f.w;
  *(uint2*)(out + i) = o.u;
}

__device__ __forceinline__ float tern1(float w) {
  float r = rintf(w);                       // round-half-even like jnp.round
  return (r > 0.f) ? 1.f : ((r < 0.f) ? -1.f : 0.f);
}

__global__ __launch_bounds__(256)
void tern_w_bf16(const float* __restrict__ in, bf16_t* __restrict__ out, long n) {
  long i = ((long)blockIdx.x * 256 + threadIdx.x) * 4;
  if (i >= n) return;
  float4 f = *(const float4*)(in + i);
  union { bf16_t b[4]; uint2 u; } o;
  o.b[0] = (bf16_t)tern1(f.x); o.b[1] = (bf16_t)tern1(f.y);
  o.b[2] = (bf16_t)tern1(f.z); o.b[3] = (bf16_t)tern1(f.w);
  *(uint2*)(out + i) = o.u;
}

// ---------------- async global -> LDS helpers (CDNA5) ----------------

__device__ __forceinline__ void async_g2l_b128(uint32_t lds_off, const void* gptr) {
  // GV mode: per-lane 16B  copy from global vaddr to LDS at vdst
  asm volatile("global_load_async_to_lds_b128 %0, %1, off"
               :: "v"(lds_off), "v"(gptr) : "memory");
}
__device__ __forceinline__ void wait_async0() {
  asm volatile("s_wait_asynccnt 0x0" ::: "memory");
}

// ---------------- WMMA GEMM ----------------
// C[N,M] = A[N,K] * W[M,K]^T   (both row-major, K contiguous -> NT gemm)
// Block tile 128x256 (8 waves, 2x4), wave tile 64x64 (16 wmma tiles), BK=32.

#define BM 128
#define BN 256
#define BK 32
#define LDT 40   // padded LDS row stride in elems (80B = 5*16B -> conflict-free b128)

template<bool RELU, typename OutT>
__global__ __launch_bounds__(256)
void gemm_tn_wmma(const bf16_t* __restrict__ A,
                  const bf16_t* __restrict__ W,
                  OutT* __restrict__ C,
                  int N, int K, int M)
{
  __shared__ __align__(16) bf16_t As[2][BM * LDT];
  __shared__ __align__(16) bf16_t Ws[2][BN * LDT];

  const int tid   = threadIdx.x;
  const int lane  = tid & 31;
  const int warp  = tid >> 5;      // 0..7
  const int warpN = warp & 3;      // 4 col groups of 64
  const int warpM = warp >> 2;     // 2 row groups of 64

  const int rowBase = blockIdx.y * BM;
  const int colBase = blockIdx.x * BN;

  // async tile fill: chunks of 8 bf16 (16B). A: 128*4 = 512 chunks (2/thread),
  // W: 256*4 = 1024 chunks (4/thread). Lanes cover 4 chunks x 8 rows (64B/row).
  auto issue_async = [&](int buf, int kt) {
    #pragma unroll
    for (int i = 0; i < 2; ++i) {
      const int c = tid + i * 256;
      const int row = c >> 2, col = (c & 3) * 8;
      const uint32_t loff = (uint32_t)(uintptr_t)&As[buf][row * LDT + col];
      const bf16_t* g = A + (size_t)(rowBase + row) * K + (size_t)kt * BK + col;
      async_g2l_b128(loff, g);
      __builtin_prefetch(g + 4 * BK, 0, 0);   // speculative, dropped if OOB
    }
    #pragma unroll
    for (int i = 0; i < 4; ++i) {
      const int c = tid + i * 256;
      const int row = c >> 2, col = (c & 3) * 8;
      const uint32_t loff = (uint32_t)(uintptr_t)&Ws[buf][row * LDT + col];
      const bf16_t* g = W + (size_t)(colBase + row) * K + (size_t)kt * BK + col;
      async_g2l_b128(loff, g);
      __builtin_prefetch(g + 4 * BK, 0, 0);
    }
  };

  v8f acc[2][4][4];  // [mt pair? no: 4 m-tiles x 4 n-tiles] -- flatten as [4][4] below
  #pragma unroll
  for (int t = 0; t < 2; ++t)
    #pragma unroll
    for (int a = 0; a < 4; ++a)
      #pragma unroll
      for (int b = 0; b < 4; ++b)
        if (t == 0) acc[0][a][b] = (v8f){0.f,0.f,0.f,0.f,0.f,0.f,0.f,0.f};

  const int KT = K / BK;

  issue_async(0, 0);
  wait_async0();
  __syncthreads();

  // ISA 16-bit A-operand layout: lane<16 holds K{0..7,16..23}, lane>=16 holds K{8..15,24..31}
  const int hi8  = (lane >> 4) * 8;
  const int mrow = lane & 15;

  for (int kt = 0; kt < KT; ++kt) {
    const int cur = kt & 1;
    const bool more = (kt + 1 < KT);
    if (more) issue_async(cur ^ 1, kt + 1);   // overlaps with wmma below

    Frag fa[4], fw[4];
    #pragma unroll
    for (int t = 0; t < 4; ++t) {
      const int r = warpM * 64 + t * 16 + mrow;
      fa[t].q[0] = *(const uint4*)&As[cur][r * LDT + hi8];
      fa[t].q[1] = *(const uint4*)&As[cur][r * LDT + 16 + hi8];
    }
    #pragma unroll
    for (int nt = 0; nt < 4; ++nt) {
      const int r = warpN * 64 + nt * 16 + mrow;
      fw[nt].q[0] = *(const uint4*)&Ws[cur][r * LDT + hi8];
      fw[nt].q[1] = *(const uint4*)&Ws[cur][r * LDT + 16 + hi8];
    }

    #pragma unroll
    for (int t = 0; t < 4; ++t)
      #pragma unroll
      for (int nt = 0; nt < 4; ++nt)
        acc[0][t][nt] = __builtin_amdgcn_wmma_f32_16x16x32_bf16(
            false, fa[t].v, false, fw[nt].v,
            (short)0, acc[0][t][nt], false, false);

    if (more) wait_async0();   // fills for tile kt+1 complete
    __syncthreads();
  }

  // epilogue: VGPR r holds M=r (lanes 0-15) / M=8+r (lanes 16-31), N=lane&15
  const int m_in = (lane >> 4) * 8;
  const int n_in = lane & 15;
  #pragma unroll
  for (int t = 0; t < 4; ++t) {
    #pragma unroll
    for (int nt = 0; nt < 4; ++nt) {
      const size_t col  = (size_t)(colBase + warpN * 64 + nt * 16 + n_in);
      const size_t row0 = (size_t)(rowBase + warpM * 64 + t * 16 + m_in);
      #pragma unroll
      for (int r = 0; r < 8; ++r) {
        float v = acc[0][t][nt][r];
        if (RELU) v = fmaxf(v, 0.f);
        C[(row0 + r) * (size_t)M + col] = (OutT)v;
      }
    }
  }
}

// ---------------- launch ----------------

extern "C" void kernel_launch(void* const* d_in, const int* in_sizes, int n_in,
                              void* d_out, int out_size, void* d_ws, size_t ws_size,
                              hipStream_t stream) {
  (void)in_sizes; (void)n_in; (void)out_size; (void)ws_size;

  const float* x  = (const float*)d_in[0];
  const float* W1 = (const float*)d_in[1];
  const float* W2 = (const float*)d_in[2];
  const float* W3 = (const float*)d_in[3];
  float* out = (float*)d_out;

  const long Nn = 16384, D0 = 2048, D1 = 8192, D2 = 8192, D3 = 2048;

  char* p = (char*)d_ws;
  bf16_t* xb  = (bf16_t*)p; p += (size_t)Nn * D0 * 2;   //  64 MB
  bf16_t* w1b = (bf16_t*)p; p += (size_t)D1 * D0 * 2;   //  32 MB
  bf16_t* w2b = (bf16_t*)p; p += (size_t)D2 * D1 * 2;   // 128 MB
  bf16_t* w3b = (bf16_t*)p; p += (size_t)D3 * D2 * 2;   //  32 MB
  bf16_t* h1  = (bf16_t*)p; p += (size_t)Nn * D1 * 2;   // 256 MB
  bf16_t* h2  = (bf16_t*)p;                              // 256 MB

  auto blocks = [](long n) { return (unsigned)((n + 1023) / 1024); };

  cvt_x_bf16 <<<blocks(Nn * D0), 256, 0, stream>>>(x,  xb,  Nn * D0);
  tern_w_bf16<<<blocks(D1 * D0), 256, 0, stream>>>(W1, w1b, D1 * D0);
  tern_w_bf16<<<blocks(D2 * D1), 256, 0, stream>>>(W2, w2b, D2 * D1);
  tern_w_bf16<<<blocks(D3 * D2), 256, 0, stream>>>(W3, w3b, D3 * D2);

  gemm_tn_wmma<true,  bf16_t><<<dim3(D1 / BN, Nn / BM), 256, 0, stream>>>(xb, w1b, h1, Nn, D0, D1);
  gemm_tn_wmma<true,  bf16_t><<<dim3(D2 / BN, Nn / BM), 256, 0, stream>>>(h1, w2b, h2, Nn, D1, D2);
  gemm_tn_wmma<false, float ><<<dim3(D3 / BN, Nn / BM), 256, 0, stream>>>(h2, w3b, out, Nn, D2, D3);
}
